// SlotAttentionEncoder_11269994184736
// MI455X (gfx1250) — matrix operand order repaired and checked by reading, not compiled, not run on hardware
//
#include <hip/hip_runtime.h>
#include <hip/hip_bf16.h>

// ---------------------------------------------------------------------------
// CDNA5 (gfx1250) slot-attention encoder.
// All heavy math (convs, projections, attention einsums) runs on
// v_wmma_f32_16x16x32_f16 (wave32 WMMA), f16 data / f32 accumulate.
// Convs are software-pipelined: global fetch of tile p+1 is issued before the
// WMMA stream for tile p, staged registers drain to LDS behind the barrier.
// ---------------------------------------------------------------------------

typedef __attribute__((ext_vector_type(16))) _Float16 v16h;
typedef __attribute__((ext_vector_type(8)))  float    v8f;

#define BB   32
#define KS   7
#define QD   64
#define HID  128
#define HH   128
#define WW   128
#define NPIX (HH * WW)          // 16384 pixels per image

union V16HU { v16h v; uint4 q[2]; };

__device__ __forceinline__ v8f zero8() {
    v8f z = {0.f, 0.f, 0.f, 0.f, 0.f, 0.f, 0.f, 0.f};
    return z;
}

// A fragment, 16x32 f16 (ISA 7.12.2): lane l -> row = row0 + (l&15);
// lanes 0-15 hold K = kb..kb+7 (V0-3) and kb+16..kb+23 (V4-7) with kb=0,
// lanes 16-31 with kb=8.  Two contiguous 16B reads per lane.
__device__ __forceinline__ v16h load_frag_a(const _Float16* base, int row0,
                                            int k0, int ld) {
    int l  = threadIdx.x & 31;
    int m  = l & 15;
    int kb = (l < 16) ? 0 : 8;
    const _Float16* p = base + (row0 + m) * ld + k0 + kb;
    V16HU r;
    r.q[0] = *(const uint4*)(p);
    r.q[1] = *(const uint4*)(p + 16);
    return r.v;
}

// B fragment, 32x16 f16, loaded from TRANSPOSED storage T[col][k]:
// lane l -> col = col0 + (l&15); K = kb..kb+7 and kb+16..kb+23, kb = (l<16)?0:8.
__device__ __forceinline__ v16h load_frag_b(const _Float16* baseT, int col0,
                                            int k0, int ld) {
    int l  = threadIdx.x & 31;
    int n  = l & 15;
    int kb = (l < 16) ? 0 : 8;
    const _Float16* p = baseT + (col0 + n) * ld + k0 + kb;
    V16HU r;
    r.q[0] = *(const uint4*)(p);
    r.q[1] = *(const uint4*)(p + 16);
    return r.v;
}

__device__ __forceinline__ v8f wmma16(v16h a, v16h b, v8f c) {
    return __builtin_amdgcn_wmma_f32_16x16x32_f16(false, a, false, b,
                                                  (short)0, c, false, false);
}

// ---------------------------------------------------------------------------
// Packing kernels (layout transforms, run every call - no cached state)
// ---------------------------------------------------------------------------

// data [B,5,H,W] f32 -> f16 NHWC with C padded to 32; one pixel per thread,
// 4x 16B stores.
__global__ void pack_input_kernel(const float* __restrict__ data,
                                  _Float16* __restrict__ out) {
    long pix = (long)blockIdx.x * 256 + threadIdx.x;   // over B*H*W
    if (pix >= (long)BB * NPIX) return;
    int x = (int)(pix & 127), y = (int)((pix >> 7) & 127), b = (int)(pix >> 14);
    union { uint4 u[4]; _Float16 h[32]; } pk;
#pragma unroll
    for (int c = 0; c < 32; ++c) pk.h[c] = (_Float16)0.f;
#pragma unroll
    for (int c = 0; c < 5; ++c)
        pk.h[c] = (_Float16)data[(((long)b * 5 + c) * HH + y) * WW + x];
#pragma unroll
    for (int i = 0; i < 4; ++i)
        *(uint4*)(out + pix * 32 + i * 8) = pk.u[i];
}

// conv1_w [128,5,5,5] OIHW -> WT[o][k], k = (ky*5+kx)*32 + c (c>=5 zero)
__global__ void pack_w1_kernel(const float* __restrict__ w,
                               _Float16* __restrict__ wT) {
    int i = blockIdx.x * 256 + threadIdx.x;          // 128*800
    if (i >= 128 * 800) return;
    int k = i % 800, o = i / 800;
    int c = k & 31, p = k >> 5;
    float v = 0.f;
    if (c < 5) v = w[(o * 5 + c) * 25 + p];
    wT[i] = (_Float16)v;
}

// conv{2,3}_w [128,128,5,5] OIHW -> WT[o][k], k = (ky*5+kx)*128 + c
__global__ void pack_w_kernel(const float* __restrict__ w,
                              _Float16* __restrict__ wT) {
    int i = blockIdx.x * 256 + threadIdx.x;          // 128*3200
    if (i >= 128 * 3200) return;
    int k = i % 3200, o = i / 3200;
    int c = k & 127, p = k >> 7;
    wT[i] = (_Float16)w[(o * 128 + c) * 25 + p];
}

// toK_w/toV_w [64,128] -> WT[n][k], n<64 keys, n>=64 vals
__global__ void pack_wp_kernel(const float* __restrict__ kw,
                               const float* __restrict__ vw,
                               _Float16* __restrict__ wT) {
    int i = blockIdx.x * 256 + threadIdx.x;          // 128*128
    if (i >= 128 * 128) return;
    int k = i & 127, n = i >> 7;
    wT[i] = (_Float16)(n < 64 ? kw[n * 128 + k] : vw[(n - 64) * 128 + k]);
}

// ---------------------------------------------------------------------------
// Implicit-GEMM 5x5 conv, pad 2, stride 1, NHWC f16, 128 out channels.
// Workgroup: 8 waves; tile = 64 pixels (one half-row) x 128 out channels.
// Wave w -> out channels [16w,16w+16), 4 M-tiles of 16 pixels.
// Register-staged software pipeline over the 25 filter positions.
// ---------------------------------------------------------------------------
template <int CIN, bool RELU>
__global__ __launch_bounds__(256) void conv_kernel(
    const _Float16* __restrict__ in,   // [B][H][W][CIN]
    const _Float16* __restrict__ wT,   // [128][25*CIN]
    const float*    __restrict__ bias, // [128]
    _Float16*       __restrict__ out)  // [B][H][W][128]
{
    constexpr int H8 = CIN / 8;            // 16B chunks per row
    constexpr int NA = 64 * H8 / 256;      // A chunks per thread (4 / 1)
    constexpr int NB = 128 * H8 / 256;     // B chunks per thread (8 / 2)
    __shared__ _Float16 As[64 * CIN];
    __shared__ _Float16 Bs[128 * CIN];
    const int tid  = threadIdx.x;
    const int wave = tid >> 5;
    const int wg   = blockIdx.x;           // 32*128*2
    const int x0   = (wg & 1) * 64;
    const int y    = (wg >> 1) & 127;
    const int b    = wg >> 8;
    const int Ktot = 25 * CIN;

    int apix[NA], ac8[NA], bo[NB], bc8[NB];
#pragma unroll
    for (int i = 0; i < NA; ++i) {
        int idx = tid + i * 256;
        apix[i] = idx / H8;
        ac8[i]  = idx % H8;
    }
#pragma unroll
    for (int i = 0; i < NB; ++i) {
        int idx = tid + i * 256;
        bo[i]  = idx / H8;
        bc8[i] = idx % H8;
    }

    uint4 ra[NA], rb[NB];
    auto fetch = [&](int p) {              // issue global loads for tile p
        const int ky = p / 5, kx = p % 5;
        const int iy = y + ky - 2;
#pragma unroll
        for (int i = 0; i < NA; ++i) {
            int ix = x0 + apix[i] + kx - 2;
            uint4 v = {0u, 0u, 0u, 0u};
            if ((unsigned)iy < (unsigned)HH && (unsigned)ix < (unsigned)WW)
                v = *(const uint4*)(in + (((long)b * HH + iy) * WW + ix) * CIN +
                                    ac8[i] * 8);
            ra[i] = v;
        }
#pragma unroll
        for (int i = 0; i < NB; ++i)
            rb[i] = *(const uint4*)(wT + (long)bo[i] * Ktot + p * CIN +
                                    bc8[i] * 8);
    };
    auto stage = [&]() {                   // drain registers -> LDS
#pragma unroll
        for (int i = 0; i < NA; ++i)
            *(uint4*)(&As[apix[i] * CIN + ac8[i] * 8]) = ra[i];
#pragma unroll
        for (int i = 0; i < NB; ++i)
            *(uint4*)(&Bs[bo[i] * CIN + bc8[i] * 8]) = rb[i];
    };

    v8f acc[4];
#pragma unroll
    for (int i = 0; i < 4; ++i) acc[i] = zero8();

    fetch(0);
    stage();
    __syncthreads();
    for (int p = 0; p < 25; ++p) {
        if (p < 24) fetch(p + 1);          // loads in flight during WMMA
        const int o0 = wave * 16;
#pragma unroll
        for (int kc = 0; kc < CIN / 32; ++kc) {
            v16h bf = load_frag_b(Bs, o0, kc * 32, CIN);
#pragma unroll
            for (int mt = 0; mt < 4; ++mt) {
                v16h af = load_frag_a(As, mt * 16, kc * 32, CIN);
                acc[mt] = wmma16(af, bf, acc[mt]);
            }
        }
        __syncthreads();                   // everyone done reading LDS
        if (p < 24) {
            stage();
            __syncthreads();               // tile p+1 visible
        }
    }

    // store: D layout -> lane l: col = l&15, rows = vgpr + (l<16?0:8)
    const int l     = tid & 31;
    const int o     = wave * 16 + (l & 15);
    const int rbase = (l < 16) ? 0 : 8;
    const float bv  = bias[o];
#pragma unroll
    for (int mt = 0; mt < 4; ++mt) {
#pragma unroll
        for (int v = 0; v < 8; ++v) {
            int m = mt * 16 + rbase + v;
            float r = acc[mt][v] + bv;
            if (RELU) r = fmaxf(r, 0.f);
            out[(((long)b * HH + y) * WW + x0 + m) * 128 + o] = (_Float16)r;
        }
    }
}

// ---------------------------------------------------------------------------
// LayerNorm(enc) then fused keys/vals projection (N = 64+64) via WMMA.
// Workgroup = 16 pixels; wave w -> output cols [16w,16w+16).
// Waves 0-3 produce keys [b*N+n][64]; waves 4-7 produce valsT[b*64+q][16384]
// (wave-uniform branch, bias hoisted, vals store packed to 16B).
// ---------------------------------------------------------------------------
__global__ __launch_bounds__(256) void ln_proj_kernel(
    const _Float16* __restrict__ enc,   // [B*NPIX][128] f16
    const _Float16* __restrict__ wT,    // [128][128]
    const float* __restrict__ g, const float* __restrict__ bt,
    const float* __restrict__ kb, const float* __restrict__ vb,
    _Float16* __restrict__ keys, _Float16* __restrict__ valsT) {
    __shared__ _Float16 An[16 * 128];
    __shared__ _Float16 Bs[128 * 128];
    __shared__ float red[16][16][2];
    __shared__ float mv[16][2];
    const int tid = threadIdx.x, wave = tid >> 5;
    const int wg   = blockIdx.x;               // 32*1024
    const int b    = wg >> 10;
    const int pix0 = (wg & 1023) * 16;
    const long base = ((long)b * NPIX + pix0) * 128;

    const int pixel = tid >> 4, part = tid & 15;
    uint4 raw = *(const uint4*)(enc + base + pixel * 128 + part * 8);
    const _Float16* hp = (const _Float16*)&raw;
    float xv[8], s = 0.f, ss = 0.f;
#pragma unroll
    for (int i = 0; i < 8; ++i) {
        xv[i] = (float)hp[i];
        s += xv[i];
        ss += xv[i] * xv[i];
    }
    red[pixel][part][0] = s;
    red[pixel][part][1] = ss;
    __syncthreads();
    if (tid < 16) {
        float m = 0.f, q2 = 0.f;
        for (int i = 0; i < 16; ++i) { m += red[tid][i][0]; q2 += red[tid][i][1]; }
        m *= (1.f / 128.f);
        q2 = q2 * (1.f / 128.f) - m * m;
        mv[tid][0] = m;
        mv[tid][1] = rsqrtf(q2 + 1e-5f);
    }
    __syncthreads();
    {
        float m = mv[pixel][0], rs = mv[pixel][1];
#pragma unroll
        for (int i = 0; i < 8; ++i) {
            int c = part * 8 + i;
            An[pixel * 128 + c] = (_Float16)((xv[i] - m) * rs * g[c] + bt[c]);
        }
    }
    for (int idx = tid; idx < 128 * 16; idx += 256)
        *(uint4*)(&Bs[idx * 8]) = *(const uint4*)(wT + idx * 8);
    __syncthreads();

    v8f acc = zero8();
    const int o0 = wave * 16;
#pragma unroll
    for (int kc = 0; kc < 4; ++kc) {
        v16h af = load_frag_a(An, 0, kc * 32, 128);
        v16h bf = load_frag_b(Bs, o0, kc * 32, 128);
        acc = wmma16(af, bf, acc);
    }
    const int l = tid & 31;
    const int lane16 = l & 15;
    const int rbase = (l < 16) ? 0 : 8;
    if (wave < 4) {                            // uniform per wave: keys
        const int col = wave * 16 + lane16;
        const float bv = kb[col];
#pragma unroll
        for (int v = 0; v < 8; ++v)
            keys[((long)b * NPIX + pix0 + rbase + v) * 64 + col] =
                (_Float16)(acc[v] + bv);
    } else {                                   // uniform per wave: valsT
        const int qd = (wave - 4) * 16 + lane16;
        const float bv = vb[qd];
        union { uint4 u; _Float16 h[8]; } pk;
#pragma unroll
        for (int v = 0; v < 8; ++v) pk.h[v] = (_Float16)(acc[v] + bv);
        *(uint4*)(valsT + ((long)b * 64 + qd) * NPIX + pix0 + rbase) = pk.u;
    }
}

// vals_sum[b][q] = sum_n vals  (iteration-invariant part of the 1e-8 term)
__global__ __launch_bounds__(256) void vals_sum_kernel(
    const _Float16* __restrict__ valsT, float* __restrict__ out) {
    const int b = blockIdx.x, jq = blockIdx.y;
    const _Float16* p = valsT + ((long)b * 64 + jq) * NPIX;
    float s = 0.f;
    for (int n8 = threadIdx.x; n8 < NPIX / 8; n8 += 256) {
        uint4 raw = *(const uint4*)(p + n8 * 8);
        const _Float16* hp = (const _Float16*)&raw;
#pragma unroll
        for (int i = 0; i < 8; ++i) s += (float)hp[i];
    }
    __shared__ float r[256];
    r[threadIdx.x] = s;
    __syncthreads();
    for (int st = 128; st > 0; st >>= 1) {
        if (threadIdx.x < st) r[threadIdx.x] += r[threadIdx.x + st];
        __syncthreads();
    }
    if (threadIdx.x == 0) out[b * 64 + jq] = r[0];
}

// ---------------------------------------------------------------------------
// Attention: one iteration. Block = (b, 256-pixel chunk), 8 waves x 32 n.
//  att = LN(q) @ keys^T  (WMMA, slots padded 7->16 with zero rows)
//  softmax over slots is per-lane (each n column lives in one lane)
//  U += p @ vals (WMMA via LDS-staged p), deterministic 2-stage reduction.
// ---------------------------------------------------------------------------
__global__ __launch_bounds__(256) void attn_kernel(
    const float* __restrict__ q_in,               // [32][7][64]
    const float* __restrict__ qg, const float* __restrict__ qb,
    const _Float16* __restrict__ keys, const _Float16* __restrict__ valsT,
    float* __restrict__ Upart,                    // [32*64][7*64]
    float* __restrict__ Spart) {                  // [32*64][7]
    __shared__ _Float16 qnA[16 * 64];
    __shared__ _Float16 pA[16 * 256];
    __shared__ float mvq[KS][2];
    __shared__ float uW[8][KS * 64];
    __shared__ float sP[8][KS][16];
    const int tid = threadIdx.x, wave = tid >> 5, l = tid & 31;
    const int b  = blockIdx.x >> 6;
    const int n0 = (blockIdx.x & 63) * 256;

    for (int i = tid; i < 16 * 64; i += 256) qnA[i] = (_Float16)0.f;
    for (int i = tid; i < 16 * 256; i += 256) pA[i] = (_Float16)0.f;
    if (tid < KS) {
        float s = 0.f, ss = 0.f;
        for (int k = 0; k < 64; ++k) {
            float x = q_in[(b * KS + tid) * 64 + k];
            s += x; ss += x * x;
        }
        float m = s / 64.f, v = ss / 64.f - m * m;
        mvq[tid][0] = m;
        mvq[tid][1] = rsqrtf(v + 1e-5f);
    }
    __syncthreads();
    for (int i = tid; i < KS * 64; i += 256) {
        int srow = i >> 6, k = i & 63;
        qnA[srow * 64 + k] = (_Float16)(
            (q_in[(b * KS + srow) * 64 + k] - mvq[srow][0]) * mvq[srow][1] *
                qg[k] + qb[k]);
    }
    __syncthreads();

    const int nw = n0 + wave * 32;
    float pv[2][KS];
#pragma unroll
    for (int t = 0; t < 2; ++t) {
        v8f acc = zero8();
#pragma unroll
        for (int kc = 0; kc < 2; ++kc) {
            v16h af = load_frag_a(qnA, 0, kc * 32, 64);
            v16h bf = load_frag_b(keys + ((long)b * NPIX + nw + t * 16) * 64, 0,
                                  kc * 32, 64);
            acc = wmma16(af, bf, acc);
        }
        if (l < 16) {                              // rows 0..6 live here
            float a[KS], mx = -1e30f;
#pragma unroll
            for (int s2 = 0; s2 < KS; ++s2) {
                a[s2] = acc[s2] * 1.25f;           // (Q^-0.5)*10
                mx = fmaxf(mx, a[s2]);
            }
            float sum = 0.f;
#pragma unroll
            for (int s2 = 0; s2 < KS; ++s2) {
                a[s2] = __expf(a[s2] - mx);
                sum += a[s2];
            }
            float inv = 1.f / sum;
#pragma unroll
            for (int s2 = 0; s2 < KS; ++s2) {
                float p = a[s2] * inv;
                pv[t][s2] = p;
                pA[s2 * 256 + wave * 32 + t * 16 + l] = (_Float16)p;
            }
        }
    }
    __syncthreads();

    // U partial: A = p (rows 7..15 zero), B = valsT rows (contiguous in n)
    v16h af = load_frag_a(pA, 0, wave * 32, 256);
    v8f au[4];
#pragma unroll
    for (int qt = 0; qt < 4; ++qt) {
        v16h bf = load_frag_b(valsT + (long)b * 64 * NPIX, qt * 16, nw, NPIX);
        au[qt] = wmma16(af, bf, zero8());
    }
    if (l < 16) {
#pragma unroll
        for (int qt = 0; qt < 4; ++qt)
#pragma unroll
            for (int v = 0; v < KS; ++v)
                uW[wave][v * 64 + qt * 16 + l] = au[qt][v];
#pragma unroll
        for (int s2 = 0; s2 < KS; ++s2)
            sP[wave][s2][l] = pv[0][s2] + pv[1][s2];
    }
    __syncthreads();
    const long ci = blockIdx.x;                    // b*64 + chunk
    for (int i = tid; i < KS * 64; i += 256) {
        float s = 0.f;
#pragma unroll
        for (int w2 = 0; w2 < 8; ++w2) s += uW[w2][i];
        Upart[ci * (KS * 64) + i] = s;
    }
    if (tid < KS) {
        float s = 0.f;
        for (int w2 = 0; w2 < 8; ++w2)
            for (int i2 = 0; i2 < 16; ++i2) s += sP[w2][tid][i2];
        Spart[ci * KS + tid] = s;
    }
}

// Deterministic cross-chunk reduction of U and S
__global__ __launch_bounds__(256) void reduce_US_kernel(
    const float* __restrict__ Upart, const float* __restrict__ Spart,
    float* __restrict__ U, float* __restrict__ S) {
    int i = blockIdx.x * 256 + threadIdx.x;
    if (i < BB * KS * 64) {
        int b = i / (KS * 64), sj = i % (KS * 64);
        float s = 0.f;
        for (int ch = 0; ch < 64; ++ch)
            s += Upart[((long)(b * 64 + ch)) * (KS * 64) + sj];
        U[i] = s;
    }
    if (i < BB * KS) {
        int b = i / KS, s7 = i % KS;
        float s = 0.f;
        for (int ch = 0; ch < 64; ++ch) s += Spart[(b * 64 + ch) * KS + s7];
        S[i] = s;
    }
}

// ---------------------------------------------------------------------------
// Per-slot update: upd = U/S + 1e-8*vals_sum ; GRU ; LN ; FF ; q += ff.
// One block per (b,slot), 64 threads. Negligible FLOPs -> plain VALU.
// ---------------------------------------------------------------------------
__global__ __launch_bounds__(64) void slot_update_kernel(
    const float* __restrict__ q_in, const float* __restrict__ U,
    const float* __restrict__ S, const float* __restrict__ vals_sum,
    const float* __restrict__ wih, const float* __restrict__ whh,
    const float* __restrict__ bih, const float* __restrict__ bhh,
    const float* __restrict__ ug, const float* __restrict__ ub,
    const float* __restrict__ f1w, const float* __restrict__ f1b,
    const float* __restrict__ f2w, const float* __restrict__ f2b,
    const float* __restrict__ fw, const float* __restrict__ fb,
    float* __restrict__ q_out, float* __restrict__ dout, int write_out) {
    const int bs = blockIdx.x;                     // 0..223
    const int b = bs / KS;
    const int j = threadIdx.x;                     // 0..63
    __shared__ float upd[64], qold[64], h[64], lnh[64], ff1[16], tmp[64];
    __shared__ float mstat[2];
    upd[j]  = U[bs * 64 + j] / S[bs] + 1e-8f * vals_sum[b * 64 + j];
    float qo = q_in[bs * 64 + j];
    qold[j] = qo;
    __syncthreads();
    float gi[3], gh[3];
#pragma unroll
    for (int g = 0; g < 3; ++g) {
        int row = g * 64 + j;
        float a = bih[row], c = bhh[row];
        for (int k = 0; k < 64; ++k) {
            a += upd[k] * wih[row * 64 + k];
            c += qold[k] * whh[row * 64 + k];
        }
        gi[g] = a;
        gh[g] = c;
    }
    float r = 1.f / (1.f + __expf(-(gi[0] + gh[0])));
    float z = 1.f / (1.f + __expf(-(gi[1] + gh[1])));
    float n = tanhf(gi[2] + r * gh[2]);
    float hv = (1.f - z) * n + z * qo;
    h[j] = hv;
    __syncthreads();
    if (j == 0) {
        float s = 0.f, ss = 0.f;
        for (int k = 0; k < 64; ++k) { s += h[k]; ss += h[k] * h[k]; }
        float m = s / 64.f;
        mstat[0] = m;
        mstat[1] = rsqrtf(ss / 64.f - m * m + 1e-5f);
    }
    __syncthreads();
    lnh[j] = (hv - mstat[0]) * mstat[1] * ug[j] + ub[j];
    __syncthreads();
    if (j < 16) {
        float a = f1b[j];
        for (int k = 0; k < 64; ++k) a += lnh[k] * f1w[j * 64 + k];
        ff1[j] = fmaxf(a, 0.f);
    }
    __syncthreads();
    float a2 = f2b[j];
    for (int k = 0; k < 16; ++k) a2 += ff1[k] * f2w[j * 16 + k];
    float qn2 = hv + a2;
    q_out[bs * 64 + j] = qn2;
    if (write_out) {
        dout[BB * KS * 2 + bs * 64 + j] = qn2;     // q output after positions
        tmp[j] = qn2;
        __syncthreads();
        if (j < 2) {
            float a = fb[j];
            for (int k = 0; k < 64; ++k) a += tmp[k] * fw[j * 64 + k];
            dout[bs * 2 + j] = a;                  // positions
        }
    }
}

// ---------------------------------------------------------------------------
extern "C" void kernel_launch(void* const* d_in, const int* in_sizes, int n_in,
                              void* d_out, int out_size, void* d_ws,
                              size_t ws_size, hipStream_t stream) {
    const float* data      = (const float*)d_in[0];
    const float* slots     = (const float*)d_in[1];
    const float* c1w = (const float*)d_in[2],  *c1b = (const float*)d_in[3];
    const float* c2w = (const float*)d_in[4],  *c2b = (const float*)d_in[5];
    const float* c3w = (const float*)d_in[6],  *c3b = (const float*)d_in[7];
    const float* dNg = (const float*)d_in[8],  *dNb = (const float*)d_in[9];
    const float* qNg = (const float*)d_in[10], *qNb = (const float*)d_in[11];
    const float* uNg = (const float*)d_in[12], *uNb = (const float*)d_in[13];
    const float* tKw = (const float*)d_in[14], *tKb = (const float*)d_in[15];
    const float* tVw = (const float*)d_in[16], *tVb = (const float*)d_in[17];
    const float* wih = (const float*)d_in[18], *whh = (const float*)d_in[19];
    const float* bih = (const float*)d_in[20], *bhh = (const float*)d_in[21];
    const float* f1w = (const float*)d_in[22], *f1b = (const float*)d_in[23];
    const float* f2w = (const float*)d_in[24], *f2b = (const float*)d_in[25];
    const float* fw  = (const float*)d_in[26], *fb  = (const float*)d_in[27];

    char* base = (char*)d_ws;
    size_t off = 0;
    auto take = [&](size_t bytes) {
        void* p = base + off;
        off += (bytes + 255) & ~(size_t)255;
        return p;
    };
    _Float16* xin   = (_Float16*)take((size_t)BB * NPIX * 32 * 2);
    _Float16* buf1  = (_Float16*)take((size_t)BB * NPIX * 128 * 2);
    _Float16* buf2  = (_Float16*)take((size_t)BB * NPIX * 128 * 2);
    _Float16* wt1   = (_Float16*)take((size_t)128 * 800 * 2);
    _Float16* wt2   = (_Float16*)take((size_t)128 * 3200 * 2);
    _Float16* wt3   = (_Float16*)take((size_t)128 * 3200 * 2);
    _Float16* wtp   = (_Float16*)take((size_t)128 * 128 * 2);
    _Float16* keys  = (_Float16*)take((size_t)BB * NPIX * 64 * 2);
    _Float16* valsT = (_Float16*)take((size_t)BB * 64 * NPIX * 2);
    float* vsum  = (float*)take((size_t)BB * 64 * 4);
    float* qbuf  = (float*)take((size_t)BB * KS * QD * 4);
    float* Upart = (float*)take((size_t)BB * 64 * KS * 64 * 4);
    float* Spart = (float*)take((size_t)BB * 64 * KS * 4);
    float* Ubuf  = (float*)take((size_t)BB * KS * 64 * 4);
    float* Sbuf  = (float*)take((size_t)BB * KS * 4);

    // ---- pack inputs / weights into WMMA-friendly f16 layouts ----
    pack_input_kernel<<<(BB * NPIX + 255) / 256, 256, 0, stream>>>(data, xin);
    pack_w1_kernel<<<(128 * 800 + 255) / 256, 256, 0, stream>>>(c1w, wt1);
    pack_w_kernel<<<(128 * 3200 + 255) / 256, 256, 0, stream>>>(c2w, wt2);
    pack_w_kernel<<<(128 * 3200 + 255) / 256, 256, 0, stream>>>(c3w, wt3);
    pack_wp_kernel<<<(128 * 128 + 255) / 256, 256, 0, stream>>>(tKw, tVw, wtp);

    // ---- CNN encoder (implicit-GEMM WMMA convs, software pipelined) ----
    conv_kernel<32, true><<<BB * HH * 2, 256, 0, stream>>>(xin, wt1, c1b, buf1);
    conv_kernel<128, true><<<BB * HH * 2, 256, 0, stream>>>(buf1, wt2, c2b, buf2);
    conv_kernel<128, false><<<BB * HH * 2, 256, 0, stream>>>(buf2, wt3, c3b, buf1);

    // ---- LN + keys/vals projection, vals column-sum ----
    ln_proj_kernel<<<BB * (NPIX / 16), 256, 0, stream>>>(buf1, wtp, dNg, dNb,
                                                         tKb, tVb, keys, valsT);
    vals_sum_kernel<<<dim3(BB, 64), 256, 0, stream>>>(valsT, vsum);

    // ---- 3 slot-attention iterations ----
    const float* qcur = slots;
    for (int it = 0; it < 3; ++it) {
        attn_kernel<<<BB * 64, 256, 0, stream>>>(qcur, qNg, qNb, keys, valsT,
                                                 Upart, Spart);
        reduce_US_kernel<<<(BB * KS * 64 + 255) / 256, 256, 0, stream>>>(
            Upart, Spart, Ubuf, Sbuf);
        slot_update_kernel<<<BB * KS, 64, 0, stream>>>(
            qcur, Ubuf, Sbuf, vsum, wih, whh, bih, bhh, uNg, uNb, f1w, f1b,
            f2w, f2b, fw, fb, qbuf, (float*)d_out, it == 2 ? 1 : 0);
        qcur = qbuf;
    }
    (void)in_sizes; (void)n_in; (void)out_size; (void)ws_size;
}
